// MyMambaPipeline_90761248899292
// MI455X (gfx1250) — matrix operand order repaired and checked by reading, not compiled, not run on hardware
//
#include <hip/hip_runtime.h>
#include <math.h>

// ---------------- problem constants ----------------
#define B_    2
#define T_    4
#define NP_   196
#define L_    784            // T_*NP_
#define BT_   1568           // B_*L_
#define DM_   256
#define DI_   512
#define DSTATE_ 16
#define DR_   16
#define DEPTH_ 12

typedef __attribute__((ext_vector_type(16))) _Float16 v16h;
typedef __attribute__((ext_vector_type(8)))  _Float16 v8h;
typedef __attribute__((ext_vector_type(8)))  float    v8f;

// ---------------- WMMA fragment loaders (CDNA5 16x16x32 f16 layouts) ------
// A tile: 16(M) x 32(K), row-major source A[M][lda]. Lanes 0-15: M=lane,
// K runs [half*8 .. half*8+7] and [16+half*8 .. 16+half*8+7] (half = lane>>4).
__device__ inline v16h load_a_frag(const _Float16* __restrict__ A, int lda,
                                   int m0, int k0, int lane) {
    int mr = lane & 15, hf = lane >> 4;
    const _Float16* p = A + (size_t)(m0 + mr) * lda + k0 + hf * 8;
    union { v16h v; v8h h[2]; } u;
    u.h[0] = *(const v8h*)(p);
    u.h[1] = *(const v8h*)(p + 16);
    return u.v;
}
// B tile: 32(K) x 16(N) from weight W[N][ldw] (row-major, as in x @ W.T).
// Lane = column N; K runs [half*8 .. half*8+7] and [16+half*8 .. 16+half*8+7] (half = lane>>4).
__device__ inline v16h load_b_frag(const _Float16* __restrict__ W, int ldw,
                                   int n0, int k0, int lane) {
    int nr = lane & 15, hf = lane >> 4;
    const _Float16* p = W + (size_t)(n0 + nr) * ldw + k0 + hf * 8;
    union { v16h v; v8h h[2]; } u;
    u.h[0] = *(const v8h*)(p);
    u.h[1] = *(const v8h*)(p + 16);
    return u.v;
}

// ---------------- GEMM: C[M][N] = A[M][K] * W[N][K]^T  (split-f16, fp32 acc)
__global__ void gemm_wmma_kernel(const _Float16* __restrict__ Ahi,
                                 const _Float16* __restrict__ Alo,
                                 const _Float16* __restrict__ Whi,
                                 const _Float16* __restrict__ Wlo,
                                 float* __restrict__ C, int M, int N, int K) {
    int wave = (blockIdx.x * blockDim.x + threadIdx.x) >> 5;
    int lane = threadIdx.x & 31;
    int tilesN = N >> 4;
    int totTiles = (M >> 4) * tilesN;
    if (wave >= totTiles) return;                    // wave-uniform exit: EXEC stays full
    int m0 = (wave / tilesN) << 4;
    int n0 = (wave % tilesN) << 4;

    v8f acc = {};
    for (int k0 = 0; k0 < K; k0 += 32) {
        v16h ah = load_a_frag(Ahi, K, m0, k0, lane);
        v16h al = load_a_frag(Alo, K, m0, k0, lane);
        v16h bh = load_b_frag(Whi, K, n0, k0, lane);
        v16h bl = load_b_frag(Wlo, K, n0, k0, lane);
        if (k0 + 32 < K)
            __builtin_prefetch(Ahi + (size_t)(m0 + (lane & 15)) * K + k0 + 32, 0, 3);
        // 2-term split product: ah*bh + ah*bl + al*bh  (al*bl ~ 2^-22, dropped)
        acc = __builtin_amdgcn_wmma_f32_16x16x32_f16(false, al, false, bh, (short)0, acc, false, false);
        acc = __builtin_amdgcn_wmma_f32_16x16x32_f16(false, ah, false, bl, (short)0, acc, false, false);
        acc = __builtin_amdgcn_wmma_f32_16x16x32_f16(false, ah, false, bh, (short)0, acc, false, false);
    }
    // C/D layout: VGPR r -> M = r (lanes 0-15) / M = 8+r (lanes 16-31), N = lane&15
    int hf = lane >> 4, nc = lane & 15;
    #pragma unroll
    for (int r = 0; r < 8; ++r)
        C[(size_t)(m0 + r + hf * 8) * N + n0 + nc] = acc[r];
}

// ---------------- split fp32 -> (hi,lo) f16 ----------------
__global__ void split_f16_kernel(const float* __restrict__ src,
                                 _Float16* __restrict__ hi,
                                 _Float16* __restrict__ lo, int n) {
    int i = blockIdx.x * blockDim.x + threadIdx.x;
    if (i >= n) return;
    float v = src[i];
    _Float16 h = (_Float16)v;
    hi[i] = h;
    lo[i] = (_Float16)(v - (float)h);
}

// ---------------- patch embed + pos/temp embeddings ----------------
__global__ void patch_embed_kernel(const float* __restrict__ x,
                                   const float* __restrict__ pw,
                                   const float* __restrict__ pb,
                                   const float* __restrict__ pos,
                                   const float* __restrict__ tpos,
                                   float* __restrict__ tok) {
    int tkn = blockIdx.x;                 // 0..BT_-1
    int c = threadIdx.x;                  // 0..255
    int b = tkn / L_, r = tkn % L_;
    int t = r / NP_, p = r % NP_;
    int py = p / 14, px = p % 14;
    __shared__ float patch[768];
    for (int k = threadIdx.x; k < 768; k += 256) {
        int ch = k >> 8, rem = k & 255, yy = rem >> 4, xx = rem & 15;
        patch[k] = x[(((size_t)(b * 3 + ch) * T_ + t) * 224 + py * 16 + yy) * 224
                     + px * 16 + xx];
    }
    __syncthreads();
    float acc = pb[c];
    const float* wr = pw + (size_t)c * 768;
    #pragma unroll 4
    for (int k = 0; k < 768; ++k) acc += patch[k] * wr[k];
    tok[(size_t)tkn * DM_ + c] = acc + pos[p * DM_ + c] + tpos[t * DM_ + c];
}

// ---------------- residual add + RMSNorm -> split f16 ----------------
__global__ void add_rmsnorm_split_kernel(const float* __restrict__ hidden,
                                         float* __restrict__ residual,
                                         int first, const float* __restrict__ nw,
                                         _Float16* __restrict__ hi,
                                         _Float16* __restrict__ lo) {
    int tkn = blockIdx.x, c = threadIdx.x;
    size_t idx = (size_t)tkn * DM_ + c;
    float r = first ? hidden[idx] : (hidden[idx] + residual[idx]);
    residual[idx] = r;
    float ss = r * r;
    #pragma unroll
    for (int m = 16; m >= 1; m >>= 1) ss += __shfl_xor(ss, m);
    __shared__ float lds[8];
    if ((threadIdx.x & 31) == 0) lds[threadIdx.x >> 5] = ss;
    __syncthreads();
    float tot = 0.f;
    #pragma unroll
    for (int i = 0; i < 8; ++i) tot += lds[i];
    float v = r * rsqrtf(tot * (1.f / DM_) + 1e-5f) * nw[c];
    _Float16 h = (_Float16)v;
    hi[idx] = h;
    lo[idx] = (_Float16)(v - (float)h);
}

// ---------------- causal depthwise conv (4 taps) + silu, dir-ordered -------
__global__ void conv_silu_split_kernel(const float* __restrict__ xz,
                                       const float* __restrict__ cw,
                                       const float* __restrict__ cb, int dir,
                                       float* __restrict__ xs,
                                       _Float16* __restrict__ hi,
                                       _Float16* __restrict__ lo) {
    int id = blockIdx.x * blockDim.x + threadIdx.x;
    if (id >= BT_ * DI_) return;
    int d = id % DI_, tkn = id / DI_;
    int b = tkn / L_, l = tkn % L_;
    float acc = cb[d];
    #pragma unroll
    for (int k = 0; k < 4; ++k) {
        int lp = l - 3 + k;
        if (lp >= 0) {
            int src = dir ? (L_ - 1 - lp) : lp;
            acc += xz[((size_t)b * L_ + src) * (2 * DI_) + d] * cw[d * 4 + k];
        }
    }
    float s = acc / (1.f + __expf(-acc));     // silu
    xs[id] = s;
    _Float16 h = (_Float16)s;
    hi[id] = h;
    lo[id] = (_Float16)(s - (float)h);
}

// ---------------- dt = softplus(dbl[:, :16] @ dtw^T + dtb) (K=16 -> VALU) --
__global__ void dt_kernel(const float* __restrict__ dbl,
                          const float* __restrict__ dtw,
                          const float* __restrict__ dtb,
                          float* __restrict__ dt) {
    int id = blockIdx.x * blockDim.x + threadIdx.x;
    if (id >= BT_ * DI_) return;
    int d = id % DI_, tkn = id / DI_;
    const float* row = dbl + (size_t)tkn * 48;
    float acc = dtb[d];
    #pragma unroll
    for (int r = 0; r < DR_; ++r) acc += row[r] * dtw[d * DR_ + r];
    dt[id] = (acc > 20.f) ? acc : log1pf(__expf(acc));
}

// ---------------- selective scan: 16-state striped across a half-wave ------
// gid/16 = (b,d) channel pair; lane&15 = state index s.
__global__ void scan_kernel(const float* __restrict__ xs,
                            const float* __restrict__ dt,
                            const float* __restrict__ dbl,
                            const float* __restrict__ A_log,
                            const float* __restrict__ Ds,
                            const float* __restrict__ xz,
                            float* __restrict__ ysum, int dir) {
    int gid = blockIdx.x * blockDim.x + threadIdx.x;
    int pair = gid >> 4;
    if (pair >= B_ * DI_) return;
    int s = gid & 15;
    int b = pair / DI_, d = pair % DI_;
    float a = -__expf(A_log[d * DSTATE_ + s]);
    float dsd = Ds[d];
    float h = 0.f;
    for (int l = 0; l < L_; ++l) {
        size_t tk = (size_t)b * L_ + l;
        float dtv = dt[tk * DI_ + d];        // broadcast across the 16-group
        float u   = xs[tk * DI_ + d];
        float Bv  = dbl[tk * 48 + 16 + s];
        float Cv  = dbl[tk * 48 + 32 + s];
        h = __expf(dtv * a) * h + (dtv * u) * Bv;
        float p = h * Cv;
        p += __shfl_xor(p, 8);
        p += __shfl_xor(p, 4);
        p += __shfl_xor(p, 2);
        p += __shfl_xor(p, 1);
        if (s == 0) {
            int oi = dir ? (L_ - 1 - l) : l;          // un-reverse on the fly
            size_t ot = (size_t)b * L_ + oi;
            float val = p + dsd * u;
            float z = xz[ot * (2 * DI_) + DI_ + d];
            val *= z / (1.f + __expf(-z));            // * silu(z)
            if (dir) ysum[ot * DI_ + d] += val;
            else     ysum[ot * DI_ + d]  = val;
        }
    }
}

// ---------------- final residual + RMSNorm (tokens 0..4 only) --------------
__global__ void final_norm_kernel(const float* __restrict__ hidden,
                                  const float* __restrict__ residual,
                                  const float* __restrict__ nw,
                                  float* __restrict__ fus) {
    int blk = blockIdx.x;                 // 0..9
    int b = blk / 5, tk = blk % 5, c = threadIdx.x;
    size_t idx = ((size_t)b * L_ + tk) * DM_ + c;
    float r = hidden[idx] + residual[idx];
    float ss = r * r;
    #pragma unroll
    for (int m = 16; m >= 1; m >>= 1) ss += __shfl_xor(ss, m);
    __shared__ float lds[8];
    if ((threadIdx.x & 31) == 0) lds[threadIdx.x >> 5] = ss;
    __syncthreads();
    float tot = 0.f;
    #pragma unroll
    for (int i = 0; i < 8; ++i) tot += lds[i];
    fus[(size_t)blk * DM_ + c] = r * rsqrtf(tot * (1.f / DM_) + 1e-5f) * nw[c];
}

// ---------------- bbox + pix heads ----------------
__global__ void heads_kernel(const float* __restrict__ fus,
                             const float* __restrict__ bw1, const float* __restrict__ bb1,
                             const float* __restrict__ bw2, const float* __restrict__ bb2,
                             const float* __restrict__ pw1, const float* __restrict__ pb1,
                             const float* __restrict__ pw2, const float* __restrict__ pb2,
                             float* __restrict__ out) {
    int blk = blockIdx.x;                 // 0..9: b*5 + head
    int b = blk / 5, head = blk % 5;
    int j = threadIdx.x;                  // 0..255
    __shared__ float h1[256];
    const float* f = fus + (size_t)(b * 5 + head) * DM_;
    float acc;
    if (head == 0) {
        acc = bb1[j];
        const float* w = bw1 + (size_t)j * DM_;
        for (int c = 0; c < DM_; ++c) acc += f[c] * w[c];
    } else {
        int i = head - 1;
        acc = pb1[i * DM_ + j];
        const float* w = pw1 + ((size_t)i * DM_ + j) * DM_;
        for (int c = 0; c < DM_; ++c) acc += f[c] * w[c];
    }
    h1[j] = fmaxf(acc, 0.f);
    __syncthreads();
    if (head == 0) {
        if (j < 3) {
            float o = bb2[j];
            const float* w = bw2 + (size_t)j * DM_;
            for (int c = 0; c < DM_; ++c) o += h1[c] * w[c];
            out[b * 11 + j] = o;
        }
    } else {
        int i = head - 1;
        if (j < 2) {
            float o = pb2[i * 2 + j];
            const float* w = pw2 + ((size_t)i * 2 + j) * DM_;
            for (int c = 0; c < DM_; ++c) o += h1[c] * w[c];
            out[b * 11 + 3 + i * 2 + j] = 1.f / (1.f + __expf(-o));
        }
    }
}

// =====================================================================
extern "C" void kernel_launch(void* const* d_in, const int* in_sizes, int n_in,
                              void* d_out, int out_size, void* d_ws, size_t ws_size,
                              hipStream_t stream) {
    (void)in_sizes; (void)n_in; (void)out_size; (void)ws_size;
    const float* x       = (const float*)d_in[0];
    const float* patch_w = (const float*)d_in[1];
    const float* patch_b = (const float*)d_in[2];
    const float* pos_e   = (const float*)d_in[3];
    const float* temp_p  = (const float*)d_in[4];
    const float* in_proj = (const float*)d_in[5];
    const float* conv_w[2] = { (const float*)d_in[6],  (const float*)d_in[13] };
    const float* conv_b[2] = { (const float*)d_in[7],  (const float*)d_in[14] };
    const float* xproj[2]  = { (const float*)d_in[8],  (const float*)d_in[15] };
    const float* dt_w[2]   = { (const float*)d_in[9],  (const float*)d_in[16] };
    const float* dt_b[2]   = { (const float*)d_in[10], (const float*)d_in[17] };
    const float* A_log[2]  = { (const float*)d_in[11], (const float*)d_in[18] };
    const float* Ds[2]     = { (const float*)d_in[12], (const float*)d_in[19] };
    const float* out_w   = (const float*)d_in[20];
    const float* norm_w  = (const float*)d_in[21];
    const float* normf_w = (const float*)d_in[22];
    const float* bw1 = (const float*)d_in[23];
    const float* bb1 = (const float*)d_in[24];
    const float* bw2 = (const float*)d_in[25];
    const float* bb2 = (const float*)d_in[26];
    const float* pw1 = (const float*)d_in[27];
    const float* pb1 = (const float*)d_in[28];
    const float* pw2 = (const float*)d_in[29];
    const float* pb2 = (const float*)d_in[30];
    float* out = (float*)d_out;

    // ---- carve workspace (256B aligned) ----
    char* cur = (char*)d_ws;
    auto alloc = [&](size_t bytes) -> char* {
        char* p = cur;
        cur += (bytes + 255) & ~(size_t)255;
        return p;
    };
    float* tok      = (float*)alloc((size_t)BT_ * DM_ * 4);
    float* residual = (float*)alloc((size_t)BT_ * DM_ * 4);
    float* hidden   = (float*)alloc((size_t)BT_ * DM_ * 4);
    float* xz       = (float*)alloc((size_t)BT_ * 2 * DI_ * 4);
    _Float16* hn_hi = (_Float16*)alloc((size_t)BT_ * DM_ * 2);
    _Float16* hn_lo = (_Float16*)alloc((size_t)BT_ * DM_ * 2);
    float* xs[2];  _Float16* xs_hi[2]; _Float16* xs_lo[2];
    float* dbl[2]; float* dtb[2];
    for (int d = 0; d < 2; ++d) {
        xs[d]    = (float*)alloc((size_t)BT_ * DI_ * 4);
        xs_hi[d] = (_Float16*)alloc((size_t)BT_ * DI_ * 2);
        xs_lo[d] = (_Float16*)alloc((size_t)BT_ * DI_ * 2);
        dbl[d]   = (float*)alloc((size_t)BT_ * 48 * 4);
        dtb[d]   = (float*)alloc((size_t)BT_ * DI_ * 4);
    }
    float* ysum     = (float*)alloc((size_t)BT_ * DI_ * 4);
    _Float16* ys_hi = (_Float16*)alloc((size_t)BT_ * DI_ * 2);
    _Float16* ys_lo = (_Float16*)alloc((size_t)BT_ * DI_ * 2);
    float* fus      = (float*)alloc((size_t)10 * DM_ * 4);
    const size_t nInW  = (size_t)DEPTH_ * 2 * DI_ * DM_;   // 12*1024*256
    const size_t nXpW  = (size_t)DEPTH_ * 48 * DI_;        // 12*48*512
    const size_t nOutW = (size_t)DEPTH_ * DM_ * DI_;       // 12*256*512
    _Float16* inw_hi  = (_Float16*)alloc(nInW * 2);
    _Float16* inw_lo  = (_Float16*)alloc(nInW * 2);
    _Float16* xpw_hi[2], *xpw_lo[2];
    for (int d = 0; d < 2; ++d) {
        xpw_hi[d] = (_Float16*)alloc(nXpW * 2);
        xpw_lo[d] = (_Float16*)alloc(nXpW * 2);
    }
    _Float16* outw_hi = (_Float16*)alloc(nOutW * 2);
    _Float16* outw_lo = (_Float16*)alloc(nOutW * 2);

    auto split = [&](const float* src, _Float16* hi, _Float16* lo, size_t n) {
        split_f16_kernel<<<(unsigned)((n + 255) / 256), 256, 0, stream>>>(src, hi, lo, (int)n);
    };
    auto gemm = [&](const _Float16* Ah, const _Float16* Al,
                    const _Float16* Wh, const _Float16* Wl,
                    float* C, int M, int N, int K) {
        int tiles = (M / 16) * (N / 16);
        gemm_wmma_kernel<<<(tiles + 7) / 8, 256, 0, stream>>>(Ah, Al, Wh, Wl, C, M, N, K);
    };

    // ---- one-time weight splits (all layers at once) ----
    split(in_proj, inw_hi, inw_lo, nInW);
    split(xproj[0], xpw_hi[0], xpw_lo[0], nXpW);
    split(xproj[1], xpw_hi[1], xpw_lo[1], nXpW);
    split(out_w, outw_hi, outw_lo, nOutW);

    // ---- patch embedding ----
    patch_embed_kernel<<<BT_, 256, 0, stream>>>(x, patch_w, patch_b, pos_e, temp_p, tok);

    const int nElemDI = BT_ * DI_;
    const int gridDI = (nElemDI + 255) / 256;

    for (int i = 0; i < DEPTH_; ++i) {
        // residual += hidden; h = rmsnorm(residual)*norm_w  ->  f16 split
        add_rmsnorm_split_kernel<<<BT_, 256, 0, stream>>>(
            (i == 0) ? tok : hidden, residual, (i == 0) ? 1 : 0,
            norm_w + (size_t)i * DM_, hn_hi, hn_lo);

        // xz = h @ in_proj^T  [1568 x 1024], K=256
        gemm(hn_hi, hn_lo, inw_hi + (size_t)i * 2 * DI_ * DM_,
             inw_lo + (size_t)i * 2 * DI_ * DM_, xz, BT_, 2 * DI_, DM_);

        for (int dir = 0; dir < 2; ++dir) {
            conv_silu_split_kernel<<<gridDI, 256, 0, stream>>>(
                xz, conv_w[dir] + (size_t)i * DI_ * 4, conv_b[dir] + (size_t)i * DI_,
                dir, xs[dir], xs_hi[dir], xs_lo[dir]);

            // dbl = xs @ xproj^T  [1568 x 48], K=512
            gemm(xs_hi[dir], xs_lo[dir], xpw_hi[dir] + (size_t)i * 48 * DI_,
                 xpw_lo[dir] + (size_t)i * 48 * DI_, dbl[dir], BT_, 48, DI_);

            dt_kernel<<<gridDI, 256, 0, stream>>>(
                dbl[dir], dt_w[dir] + (size_t)i * DI_ * DR_,
                dt_b[dir] + (size_t)i * DI_, dtb[dir]);

            scan_kernel<<<64, 256, 0, stream>>>(
                xs[dir], dtb[dir], dbl[dir],
                A_log[dir] + (size_t)i * DI_ * DSTATE_, Ds[dir] + (size_t)i * DI_,
                xz, ysum, dir);
        }

        // hidden = (yf + yb) @ out_proj^T  [1568 x 256], K=512
        split(ysum, ys_hi, ys_lo, (size_t)nElemDI);
        gemm(ys_hi, ys_lo, outw_hi + (size_t)i * DM_ * DI_,
             outw_lo + (size_t)i * DM_ * DI_, hidden, BT_, DM_, DI_);
    }

    final_norm_kernel<<<10, 256, 0, stream>>>(hidden, residual, normf_w, fus);
    heads_kernel<<<10, 256, 0, stream>>>(fus, bw1, bb1, bw2, bb2,
                                         pw1, pb1, pw2, pb2, out);
}
